// AAFM_59562606461273
// MI455X (gfx1250) — matrix operand order repaired and checked by reading, not compiled, not run on hardware
//
#include <hip/hip_runtime.h>
#include <hip/hip_bf16.h>

#define D_MODEL 512
#define SEQ 2048
#define BATCH 8

typedef __bf16 bf16;
typedef __attribute__((ext_vector_type(16))) __bf16 v16bf;
typedef __attribute__((ext_vector_type(8)))  __bf16 v8bf;
typedef __attribute__((ext_vector_type(8)))  float  v8f;
typedef __attribute__((ext_vector_type(4)))  unsigned v4u;
typedef __attribute__((ext_vector_type(4)))  int      v4i;
typedef __attribute__((ext_vector_type(8)))  int      v8i;

#define LDK 40  // padded K-stride (elements): 80B rows, 16B-aligned chunks

#if defined(__has_builtin)
#if __has_builtin(__builtin_amdgcn_tensor_load_to_lds) && __has_builtin(__builtin_amdgcn_s_wait_tensorcnt)
#define HAVE_TDM 1
#endif
#endif
#ifndef HAVE_TDM
#define HAVE_TDM 0
#endif

__device__ __forceinline__ v8f zero8() {
    v8f z;
#pragma unroll
    for (int i = 0; i < 8; ++i) z[i] = 0.0f;
    return z;
}

__device__ __forceinline__ v8bf cvt8(float4 a, float4 b) {
    v8bf r;
    r[0] = (bf16)a.x; r[1] = (bf16)a.y; r[2] = (bf16)a.z; r[3] = (bf16)a.w;
    r[4] = (bf16)b.x; r[5] = (bf16)b.y; r[6] = (bf16)b.z; r[7] = (bf16)b.w;
    return r;
}

// A fragment (16x32 bf16, MxK), ISA 7.12.2 layout; tile row-major [row][LDK].
__device__ __forceinline__ v16bf frag_A(const bf16* tile, int rowBase) {
    int lane = threadIdx.x & 31;
    int row  = rowBase + (lane & 15);
    int half = lane >> 4;
    const bf16* p = tile + row * LDK + half * 8;
    v8bf lo = *(const v8bf*)(p);        // K = 8*half .. +8
    v8bf hi = *(const v8bf*)(p + 16);   // K = 16+8*half .. +8
    v16bf f;
#pragma unroll
    for (int i = 0; i < 8; ++i) { f[i] = lo[i]; f[i + 8] = hi[i]; }
    return f;
}

// B fragment (32x16 bf16, KxN); tileT is n-major [n][LDK] (K contiguous).
__device__ __forceinline__ v16bf frag_B(const bf16* tileT, int colBase) {
    int lane = threadIdx.x & 31;
    int n    = colBase + (lane & 15);
    int half = lane >> 4;
    const bf16* p = tileT + n * LDK + half * 8;
    v8bf lo = *(const v8bf*)(p);
    v8bf hi = *(const v8bf*)(p + 16);
    v16bf f;
#pragma unroll
    for (int i = 0; i < 8; ++i) { f[i] = lo[i]; f[i + 8] = hi[i]; }
    return f;
}

__device__ __forceinline__ v8f wmma_bf16(v16bf a, v16bf b, v8f c) {
    return __builtin_amdgcn_wmma_f32_16x16x32_bf16(false, a, false, b, (short)0, c, false, false);
}

// TDM: 2-D tile, 128 rows x 32 elems, 2-byte data, row stride 2048 elems,
// LDS pad: every 16 DWORDs (64B row) insert 4 DWORDs (16B) -> LDK=40 stride.
__device__ __forceinline__ void tdm_issue_b(const bf16* gsrc, bf16* ldsdst) {
#if HAVE_TDM
    unsigned ldsa = (unsigned)(unsigned long long)(void*)ldsdst;
    unsigned long long ga = (unsigned long long)gsrc;
    v4u g0 = { 1u,                                  // count=1 (valid user D#)
               ldsa,                                // lds_addr
               (unsigned)(ga & 0xFFFFFFFFu),
               (unsigned)((ga >> 32) & 0x01FFFFFFull) | (2u << 30) };  // type=2
    v8i g1 = { (int)((1u << 16) | (1u << 20) | (3u << 22) | (3u << 25)),
               // ^ data_size=2B | pad_enable | pad_interval=16dw | pad_amount=4dw
               (int)((2048u & 0xFFFFu) << 16),      // tensor_dim0 = 2048 (lo16)
               (int)(((2048u >> 16) & 0xFFFFu) | ((1024u & 0xFFFFu) << 16)), // dim0 hi | tensor_dim1=1024 lo
               (int)((1024u >> 16) | (32u << 16)),  // dim1 hi | tile_dim0 = 32
               (int)(128u),                         // tile_dim1 = 128, tile_dim2 = 0
               (int)(2048u),                        // tensor_dim0_stride lo32 = 2048
               0, 0 };                              // stride hi / dim1_stride = 0
    v4i gz = { 0, 0, 0, 0 };
#if defined(__clang_major__) && __clang_major__ >= 23
    v8i gz8 = { 0, 0, 0, 0, 0, 0, 0, 0 };
    __builtin_amdgcn_tensor_load_to_lds(g0, g1, gz, gz, gz8, 0);
#else
    __builtin_amdgcn_tensor_load_to_lds(g0, g1, gz, gz, 0);
#endif
#else
    (void)gsrc; (void)ldsdst;
#endif
}

// ---------------- Kernel 1: q projection + sigmoid, bf16 out -------------
__global__ __launch_bounds__(256) void proj_q_kernel(
    const float* __restrict__ X, const float* __restrict__ W,
    const float* __restrict__ bias, bf16* __restrict__ qsig)
{
    __shared__ bf16 Alds[128 * LDK];
    __shared__ bf16 Blds[128 * LDK];
    const int nblock = blockIdx.x, mblock = blockIdx.y;
    const int tid  = threadIdx.x;
    const int wid  = tid >> 5, lane = tid & 31;
    const int l15  = lane & 15, half = lane >> 4;
    const int wm = (wid & 3) * 32;
    const int wn = (wid >> 2) * 64;
    const int m0 = mblock * 128, n0 = nblock * 128;
    const int lr = tid >> 1;
    const int lc = (tid & 1) * 16;

    v8f acc[2][4];
#pragma unroll
    for (int i = 0; i < 2; ++i)
#pragma unroll
        for (int j = 0; j < 4; ++j) acc[i][j] = zero8();

    for (int kk = 0; kk < D_MODEL; kk += 32) {
        const float* ap = X + (size_t)(m0 + lr) * D_MODEL + kk + lc;
        const float* bp = W + (size_t)(n0 + lr) * D_MODEL + kk + lc;
        float4 a0 = *(const float4*)(ap);     float4 a1 = *(const float4*)(ap + 4);
        float4 a2 = *(const float4*)(ap + 8); float4 a3 = *(const float4*)(ap + 12);
        float4 b0 = *(const float4*)(bp);     float4 b1 = *(const float4*)(bp + 4);
        float4 b2 = *(const float4*)(bp + 8); float4 b3 = *(const float4*)(bp + 12);
        bf16* as = Alds + lr * LDK + lc;
        bf16* bs = Blds + lr * LDK + lc;
        *(v8bf*)(as)     = cvt8(a0, a1);
        *(v8bf*)(as + 8) = cvt8(a2, a3);
        *(v8bf*)(bs)     = cvt8(b0, b1);
        *(v8bf*)(bs + 8) = cvt8(b2, b3);
        __syncthreads();

        v16bf fa0 = frag_A(Alds, wm);
        v16bf fa1 = frag_A(Alds, wm + 16);
#pragma unroll
        for (int j = 0; j < 4; ++j) {
            v16bf fb = frag_B(Blds, wn + j * 16);
            acc[0][j] = wmma_bf16(fa0, fb, acc[0][j]);
            acc[1][j] = wmma_bf16(fa1, fb, acc[1][j]);
        }
        __syncthreads();
    }

#pragma unroll
    for (int j = 0; j < 4; ++j) {
        const int col = n0 + wn + j * 16 + l15;
        const float bb = bias[col];
#pragma unroll
        for (int i = 0; i < 2; ++i) {
#pragma unroll
            for (int r = 0; r < 8; ++r) {
                const int row = m0 + wm + i * 16 + r + 8 * half;
                float v = acc[i][j][r] + bb;
                float s = 1.0f / (1.0f + __expf(-v));
                qsig[(size_t)row * D_MODEL + col] = (bf16)s;
            }
        }
    }
}

// ---- Kernel 2: k & v projections; BmatT[n][sk] = [exp(kp)*vp | exp(kp)] --
__global__ __launch_bounds__(256) void proj_kv_kernel(
    const float* __restrict__ K, const float* __restrict__ V,
    const float* __restrict__ Wk, const float* __restrict__ bk,
    const float* __restrict__ Wv, const float* __restrict__ bv,
    bf16* __restrict__ BmatT)
{
    __shared__ bf16 Aklds[128 * LDK];
    __shared__ bf16 Avlds[128 * LDK];
    __shared__ bf16 Bklds[128 * LDK];
    __shared__ bf16 Bvlds[128 * LDK];
    const int nblock = blockIdx.x, mblock = blockIdx.y;
    const int tid = threadIdx.x;
    const int wid = tid >> 5, lane = tid & 31;
    const int l15 = lane & 15, half = lane >> 4;
    const int wm = (wid & 3) * 32;
    const int wn = (wid >> 2) * 64;
    const int m0 = mblock * 128, n0 = nblock * 128;
    const int lr = tid >> 1;
    const int lc = (tid & 1) * 16;

    v8f acck[2][4], accv[2][4];
#pragma unroll
    for (int i = 0; i < 2; ++i)
#pragma unroll
        for (int j = 0; j < 4; ++j) { acck[i][j] = zero8(); accv[i][j] = zero8(); }

    for (int kk = 0; kk < D_MODEL; kk += 32) {
        const float* akp = K  + (size_t)(m0 + lr) * D_MODEL + kk + lc;
        const float* avp = V  + (size_t)(m0 + lr) * D_MODEL + kk + lc;
        const float* bkp = Wk + (size_t)(n0 + lr) * D_MODEL + kk + lc;
        const float* bvp = Wv + (size_t)(n0 + lr) * D_MODEL + kk + lc;
        {
            float4 x0 = *(const float4*)(akp),     x1 = *(const float4*)(akp + 4);
            float4 x2 = *(const float4*)(akp + 8), x3 = *(const float4*)(akp + 12);
            bf16* s = Aklds + lr * LDK + lc;
            *(v8bf*)(s) = cvt8(x0, x1); *(v8bf*)(s + 8) = cvt8(x2, x3);
        }
        {
            float4 x0 = *(const float4*)(avp),     x1 = *(const float4*)(avp + 4);
            float4 x2 = *(const float4*)(avp + 8), x3 = *(const float4*)(avp + 12);
            bf16* s = Avlds + lr * LDK + lc;
            *(v8bf*)(s) = cvt8(x0, x1); *(v8bf*)(s + 8) = cvt8(x2, x3);
        }
        {
            float4 x0 = *(const float4*)(bkp),     x1 = *(const float4*)(bkp + 4);
            float4 x2 = *(const float4*)(bkp + 8), x3 = *(const float4*)(bkp + 12);
            bf16* s = Bklds + lr * LDK + lc;
            *(v8bf*)(s) = cvt8(x0, x1); *(v8bf*)(s + 8) = cvt8(x2, x3);
        }
        {
            float4 x0 = *(const float4*)(bvp),     x1 = *(const float4*)(bvp + 4);
            float4 x2 = *(const float4*)(bvp + 8), x3 = *(const float4*)(bvp + 12);
            bf16* s = Bvlds + lr * LDK + lc;
            *(v8bf*)(s) = cvt8(x0, x1); *(v8bf*)(s + 8) = cvt8(x2, x3);
        }
        __syncthreads();

        v16bf ak0 = frag_A(Aklds, wm), ak1 = frag_A(Aklds, wm + 16);
        v16bf av0 = frag_A(Avlds, wm), av1 = frag_A(Avlds, wm + 16);
#pragma unroll
        for (int j = 0; j < 4; ++j) {
            v16bf fbk = frag_B(Bklds, wn + j * 16);
            acck[0][j] = wmma_bf16(ak0, fbk, acck[0][j]);
            acck[1][j] = wmma_bf16(ak1, fbk, acck[1][j]);
            v16bf fbv = frag_B(Bvlds, wn + j * 16);
            accv[0][j] = wmma_bf16(av0, fbv, accv[0][j]);
            accv[1][j] = wmma_bf16(av1, fbv, accv[1][j]);
        }
        __syncthreads();
    }

    // Transposed epilogue: lane's 8 acc rows are consecutive sk -> 16B stores.
#pragma unroll
    for (int j = 0; j < 4; ++j) {
        const int col = n0 + wn + j * 16 + l15;   // feature 0..511
        const float kb = bk[col];
        const float vb = bv[col];
#pragma unroll
        for (int i = 0; i < 2; ++i) {
            const int rowbase = m0 + wm + i * 16 + 8 * half;  // flat b*SEQ+sk
            const int b  = rowbase >> 11;
            const int sk = rowbase & (SEQ - 1);
            v8bf ekv, ekk;
#pragma unroll
            for (int r = 0; r < 8; ++r) {
                float ek = __expf(acck[i][j][r] + kb);
                float ev = accv[i][j][r] + vb;
                ekv[r] = (bf16)(ek * ev);
                ekk[r] = (bf16)ek;
            }
            const size_t nbase = (size_t)b * 1024;
            *(v8bf*)(BmatT + ((nbase + col) * SEQ + sk))       = ekv;
            *(v8bf*)(BmatT + ((nbase + 512 + col) * SEQ + sk)) = ekk;
        }
    }
}

// ---- Kernel 3: exp_a @ [ekv|ek]; TDM B-tiles, double-buffered LDS -------
__global__ __launch_bounds__(256) void attn_kernel(
    const float* __restrict__ dist, const bf16* __restrict__ BmatT,
    const float* __restrict__ alpha, float* __restrict__ OutWS)
{
    __shared__ bf16 Alds[2][128 * LDK];
    __shared__ bf16 Blds[2][128 * LDK];
    const int nblock = blockIdx.x;          // 0..7  (cols of 1024)
    const int mblock = blockIdx.y;          // 0..15 (rows of 2048)
    const int b      = blockIdx.z;          // batch
    const int tid = threadIdx.x;
    const int wid = tid >> 5, lane = tid & 31;
    const int l15 = lane & 15, half = lane >> 4;
    const int wm = (wid & 3) * 32;
    const int wn = (wid >> 2) * 64;
    const int m0 = mblock * 128, n0 = nblock * 128;
    const float scale = -alpha[0] * 11.0f;  // -alpha * log2(2048)

    const int lr = tid >> 1;                // loader row 0..127
    const int lc = (tid & 1) * 16;          // k-chunk 0/16

    const float* Dbase  = dist + (size_t)b * SEQ * SEQ + (size_t)(m0 + lr) * SEQ;
    const bf16*  BTbase = BmatT + ((size_t)b * 1024 + n0) * SEQ;  // row n, stride SEQ

    v8f acc[2][4];
#pragma unroll
    for (int i = 0; i < 2; ++i)
#pragma unroll
        for (int j = 0; j < 4; ++j) acc[i][j] = zero8();

    // ---- stage helpers (inlined) ----
    auto produce_A = [&](bf16* Abuf, int kk) {
        const float* dp = Dbase + kk + lc;
        __builtin_prefetch(dp + 64, 0, 3);
        float4 d0 = *(const float4*)(dp);     float4 d1 = *(const float4*)(dp + 4);
        float4 d2 = *(const float4*)(dp + 8); float4 d3 = *(const float4*)(dp + 12);
        d0.x = __expf(scale * d0.x); d0.y = __expf(scale * d0.y);
        d0.z = __expf(scale * d0.z); d0.w = __expf(scale * d0.w);
        d1.x = __expf(scale * d1.x); d1.y = __expf(scale * d1.y);
        d1.z = __expf(scale * d1.z); d1.w = __expf(scale * d1.w);
        d2.x = __expf(scale * d2.x); d2.y = __expf(scale * d2.y);
        d2.z = __expf(scale * d2.z); d2.w = __expf(scale * d2.w);
        d3.x = __expf(scale * d3.x); d3.y = __expf(scale * d3.y);
        d3.z = __expf(scale * d3.z); d3.w = __expf(scale * d3.w);
        bf16* s = Abuf + lr * LDK + lc;
        *(v8bf*)(s) = cvt8(d0, d1); *(v8bf*)(s + 8) = cvt8(d2, d3);
    };
    auto produce_B = [&](bf16* Bbuf, int kk) {
#if HAVE_TDM
        if (wid == 0) tdm_issue_b(BTbase + kk, Bbuf);   // DMA 128x32 tile, padded
#else
        // fallback: column-major global is already n-major -> direct copy
        const bf16* gp = BTbase + (size_t)lr * SEQ + kk + lc;
        uint4 r0 = *(const uint4*)(gp);
        uint4 r1 = *(const uint4*)(gp + 8);
        bf16* s = Bbuf + lr * LDK + lc;
        *(uint4*)(s) = r0; *(uint4*)(s + 8) = r1;
#endif
    };

    // ---- prologue: stage 0 ----
    produce_A(Alds[0], 0);
    produce_B(Blds[0], 0);

    const int NITER = SEQ / 32;
    for (int it = 0; it < NITER; ++it) {
        const int buf = it & 1;
#if HAVE_TDM
        __builtin_amdgcn_s_wait_tensorcnt(0);   // B_it DMA complete (wave0; no-op others)
#endif
        __syncthreads();                        // A_it visible + B_it consumable

        if (it + 1 < NITER) {                   // overlap next-stage fill with WMMA
            produce_A(Alds[buf ^ 1], (it + 1) * 32);
            produce_B(Blds[buf ^ 1], (it + 1) * 32);
        }

        v16bf fa0 = frag_A(Alds[buf], wm);
        v16bf fa1 = frag_A(Alds[buf], wm + 16);
#pragma unroll
        for (int j = 0; j < 4; ++j) {
            v16bf fb = frag_B(Blds[buf], wn + j * 16);
            acc[0][j] = wmma_bf16(fa0, fb, acc[0][j]);
            acc[1][j] = wmma_bf16(fa1, fb, acc[1][j]);
        }
    }

    float* obase = OutWS + (size_t)b * SEQ * 1024;
#pragma unroll
    for (int j = 0; j < 4; ++j) {
        const int col = n0 + wn + j * 16 + l15;  // 0..1023
#pragma unroll
        for (int i = 0; i < 2; ++i) {
#pragma unroll
            for (int r = 0; r < 8; ++r) {
                const int row = m0 + wm + i * 16 + r + 8 * half;
                obase[(size_t)row * 1024 + col] = acc[i][j][r];
            }
        }
    }
}

// ---- Kernel 4: out = sigmoid(qp) * num / (den + 1e-9) -------------------
__global__ __launch_bounds__(256) void final_kernel(
    const bf16* __restrict__ qsig, const float* __restrict__ OutWS,
    float* __restrict__ out)
{
    const size_t idx4 = ((size_t)blockIdx.x * blockDim.x + threadIdx.x) * 4;
    const size_t row = idx4 >> 9;
    const size_t col = idx4 & 511;
    const float* np = OutWS + row * 1024 + col;
    const float* dp = np + 512;
    float4 num = *(const float4*)np;
    float4 den = *(const float4*)dp;
    const bf16* qp = qsig + idx4;
    float4 r;
    r.x = (float)qp[0] * num.x / (den.x + 1e-9f);
    r.y = (float)qp[1] * num.y / (den.y + 1e-9f);
    r.z = (float)qp[2] * num.z / (den.z + 1e-9f);
    r.w = (float)qp[3] * num.w / (den.w + 1e-9f);
    *(float4*)(out + idx4) = r;
}

extern "C" void kernel_launch(void* const* d_in, const int* in_sizes, int n_in,
                              void* d_out, int out_size, void* d_ws, size_t ws_size,
                              hipStream_t stream) {
    const float* q     = (const float*)d_in[0];
    const float* k     = (const float*)d_in[1];
    const float* v     = (const float*)d_in[2];
    const float* dist  = (const float*)d_in[3];
    const float* Wq    = (const float*)d_in[4];
    const float* bq    = (const float*)d_in[5];
    const float* Wk    = (const float*)d_in[6];
    const float* bk    = (const float*)d_in[7];
    const float* Wv    = (const float*)d_in[8];
    const float* bv    = (const float*)d_in[9];
    const float* alpha = (const float*)d_in[10];
    (void)in_sizes; (void)n_in; (void)out_size; (void)ws_size;

    char* ws = (char*)d_ws;
    bf16*  BmatT = (bf16*)(ws);                                  // 32 MiB: [8][1024][2048] bf16 (n-major)
    float* OutWS = (float*)(ws + (size_t)32 * 1024 * 1024);      // 64 MiB: [16384][1024] f32
    bf16*  qsig  = (bf16*)(ws + (size_t)96 * 1024 * 1024);       // 16 MiB: [16384][512] bf16

    proj_q_kernel <<<dim3(4, 128), 256, 0, stream>>>(q, Wq, bq, qsig);
    proj_kv_kernel<<<dim3(4, 128), 256, 0, stream>>>(k, v, Wk, bk, Wv, bv, BmatT);
    attn_kernel   <<<dim3(8, 16, 8), 256, 0, stream>>>(dist, BmatT, alpha, OutWS);
    final_kernel  <<<dim3(8192), 256, 0, stream>>>(qsig, OutWS, (float*)d_out);
}
